// DRMM_class_25486335935097
// MI455X (gfx1250) — matrix operand order repaired and checked by reading, not compiled, not run on hardware
//
#include <hip/hip_runtime.h>
#include <hip/hip_bf16.h>
#include <math.h>

#define NBINS 5
#define NODES 5
#define EMB_DIM 300
#define KPAD 320            // 300 padded to 10 chunks of K=32
#define BATCH 64
#define TQ 32
#define TD 4096

typedef __attribute__((ext_vector_type(16))) _Float16 v16h;
typedef __attribute__((ext_vector_type(8)))  _Float16 v8h;
typedef __attribute__((ext_vector_type(8)))  float    v8f;

// ---------------------------------------------------------------------------
// Kernel 1: embedding gather + L2 normalize + f32->f16 + zero-pad to KPAD.
// One wave32 per row. Rows [0, B*TD) come from `sentence` -> dn,
// rows [B*TD, B*TD+B*TQ) come from `query_sentence` -> qn.
// ---------------------------------------------------------------------------
__global__ __launch_bounds__(256) void gather_norm_kernel(
    const int* __restrict__ sentence,
    const int* __restrict__ query_sentence,
    const float* __restrict__ emb,
    _Float16* __restrict__ dn,
    _Float16* __restrict__ qn)
{
    const int wrow = (blockIdx.x * blockDim.x + threadIdx.x) >> 5;
    const int lane = threadIdx.x & 31;
    const int ND = BATCH * TD;
    const int NQ = BATCH * TQ;
    if (wrow >= ND + NQ) return;

    int token;
    _Float16* outrow;
    if (wrow < ND) {
        token  = sentence[wrow];
        outrow = dn + (size_t)wrow * KPAD;
    } else {
        const int r = wrow - ND;
        token  = query_sentence[r];
        outrow = qn + (size_t)r * KPAD;
    }
    const float* erow = emb + (size_t)token * EMB_DIM;

    float vals[10];
    float ss = 0.f;
#pragma unroll
    for (int t = 0; t < 10; ++t) {
        const int k = lane + 32 * t;
        const float v = (k < EMB_DIM) ? erow[k] : 0.f;
        vals[t] = v;
        ss += v * v;
    }
#pragma unroll
    for (int off = 16; off > 0; off >>= 1)
        ss += __shfl_xor(ss, off, 32);
    const float scale = 1.f / fmaxf(sqrtf(ss), 1e-12f);  // matches F.normalize eps
#pragma unroll
    for (int t = 0; t < 10; ++t)
        outrow[lane + 32 * t] = (_Float16)(vals[t] * scale);
}

// ---------------------------------------------------------------------------
// Kernel 2: fused  sim = Qn * Dn^T  (f16 WMMA, f32 acc)  -> masked histogram
// (fully arithmetic packed register counters -> one-shot LDS flush) -> log ->
// tiny tanh MLP -> ffw_vec[b][q].
// One workgroup (8 waves) per (batch, q-tile of 16). Each wave sweeps 32
// d-tiles; A fragments resident in VGPRs across the sweep.
// ---------------------------------------------------------------------------
__global__ __launch_bounds__(256) void gemm_hist_ffw_kernel(
    const _Float16* __restrict__ qn,
    const _Float16* __restrict__ dn,
    const int*      __restrict__ sentence,
    const float*    __restrict__ w1, const float* __restrict__ b1,
    const float*    __restrict__ w2, const float* __restrict__ b2,
    float*          __restrict__ ffw_vec)
{
    __shared__ unsigned int hist[16][8];   // [q-row in tile][bin0..4, exact=5, pad]

    const int b    = blockIdx.x >> 1;
    const int qt   = blockIdx.x & 1;
    const int tid  = threadIdx.x;
    const int wid  = tid >> 5;
    const int lane = tid & 31;
    const int half = lane >> 4;            // half-wave select
    const int l16  = lane & 15;

    if (tid < 128) ((unsigned int*)hist)[tid] = 0u;
    __syncthreads();

    // --- A fragments: 16x32 f16 per chunk, contiguous k order (same as B):
    // halves[0..7]  = Qrow[k0 + half*16 + j]
    // halves[8..15] = Qrow[k0 + half*16 + 8 + j]
    const _Float16* qrow = qn + (size_t)(b * TQ + qt * 16 + l16) * KPAD;
    v16h afrag[10];
#pragma unroll
    for (int c = 0; c < 10; ++c) {
        const int k0 = c * 32;
        const v8h lo = *(const v8h*)(qrow + k0 + half * 16);
        const v8h hi = *(const v8h*)(qrow + k0 + half * 16 + 8);
        v16h a;
#pragma unroll
        for (int j = 0; j < 8; ++j) { a[j] = lo[j]; a[j + 8] = hi[j]; }
        afrag[c] = a;
    }

    // packed per-lane counters: for each of this lane's 8 output rows,
    // 6 x 8-bit fields (bins 0..4 at bytes 0..4, exact at byte 5).
    // Each field gets at most 1 increment per tile x 32 tiles -> max 32.
    unsigned long long cnt[8];
#pragma unroll
    for (int r = 0; r < 8; ++r) cnt[r] = 0ull;

    for (int dt = wid; dt < TD / 16; dt += 8) {
        const int dcol = dt * 16 + l16;
        const _Float16* drow = dn + ((size_t)b * TD + dcol) * KPAD;

        // speculative prefetch of next tile's row (stays inside workspace;
        // unconditional to avoid an exec-mask branch around it)
        __builtin_prefetch(dn + ((size_t)b * TD + dcol + 128) * KPAD, 0, 3);

        const int token = sentence[b * TD + dcol];

        // --- B fragments: 32x16 f16, column n = l16; per half-wave the 16
        // halves are 16 consecutive K values -> contiguous 32B per chunk.
        v16h bfrag[10];
#pragma unroll
        for (int c = 0; c < 10; ++c)
            bfrag[c] = *(const v16h*)(drow + c * 32 + half * 16);

        v8f cacc = {};
#pragma unroll
        for (int c = 0; c < 10; ++c)
            cacc = __builtin_amdgcn_wmma_f32_16x16x32_f16(
                false, afrag[c], false, bfrag[c], (short)0, cacc, false, false);

        // D layout: lane holds column n = l16, rows m = half*8 + r.
        // Fully arithmetic (no ternary selecting an expensive expr -> no
        // exec-mask branches): bin = trunc(v*2.5 + 2.5), clamped; validity
        // (v < 1.0, matching cum "sim < ub") and the exact window become
        // 0/1 integers that gate the unconditional shifted increment.
        // token==0 columns contribute nothing (ref adds 1e7 to them).
        const unsigned int tmask = (unsigned int)(token != 0);
#pragma unroll
        for (int r = 0; r < 8; ++r) {
            const float v = cacc[r];
            unsigned int j = (unsigned int)(int)(v * 2.5f + 2.5f);
            j = (j > 5u) ? 5u : j;                       // v_min_u32
            const unsigned int valid = tmask & (unsigned int)(v < 1.0f);
            const unsigned int ex    = tmask & (unsigned int)(v > 0.999f)
                                             & (unsigned int)(v < 1.001f);
            cnt[r] += ((unsigned long long)valid << (8u * j))
                    + ((unsigned long long)ex << 40);
        }
    }

    // one-shot flush of packed counters into the shared histogram
#pragma unroll
    for (int r = 0; r < 8; ++r) {
        const int m = half * 8 + r;
#pragma unroll
        for (int bin = 0; bin < 6; ++bin)
            atomicAdd(&hist[m][bin],
                      (unsigned int)((cnt[r] >> (8 * bin)) & 0xffull));
    }
    __syncthreads();

    // --- log(count+1) -> Linear(6,5) -> tanh -> Linear(5,1) -> tanh
    if (tid < 16) {
        float hl[NBINS + 1];
#pragma unroll
        for (int i = 0; i < NBINS + 1; ++i)
            hl[i] = logf((float)hist[tid][i] + 1.0f);
        float outv = b2[0];
#pragma unroll
        for (int n = 0; n < NODES; ++n) {
            float acc = b1[n];
#pragma unroll
            for (int i = 0; i < NBINS + 1; ++i)
                acc += w1[n * (NBINS + 1) + i] * hl[i];
            outv += w2[n] * tanhf(acc);
        }
        ffw_vec[b * TQ + qt * 16 + tid] = tanhf(outv);
    }
}

// ---------------------------------------------------------------------------
// Kernel 3: masked softmax gate over TQ=32 (one lane per query -> one wave32
// per batch) + weighted sum + final affine.
// ---------------------------------------------------------------------------
__global__ void final_kernel(const float* __restrict__ query_idf,
                             const int*   __restrict__ query_sentence,
                             const float* __restrict__ ffw_vec,
                             const float* __restrict__ gate_w,
                             const float* __restrict__ out_w,
                             const float* __restrict__ out_b,
                             float* __restrict__ out)
{
    const int b = blockIdx.x;
    const int q = threadIdx.x;                 // TQ == warpSize == 32
    float logit = query_idf[b * TQ + q] * gate_w[0];
    if (query_sentence[b * TQ + q] == 0) logit += -1e7f;

    float m = logit;
#pragma unroll
    for (int off = 16; off > 0; off >>= 1)
        m = fmaxf(m, __shfl_xor(m, off, 32));
    const float e = expf(logit - m);
    float s = e;
#pragma unroll
    for (int off = 16; off > 0; off >>= 1)
        s += __shfl_xor(s, off, 32);

    float acc = (e / s) * ffw_vec[b * TQ + q];
#pragma unroll
    for (int off = 16; off > 0; off >>= 1)
        acc += __shfl_xor(acc, off, 32);

    if (q == 0) out[b] = acc * out_w[0] + out_b[0];
}

// ---------------------------------------------------------------------------
extern "C" void kernel_launch(void* const* d_in, const int* in_sizes, int n_in,
                              void* d_out, int out_size, void* d_ws, size_t ws_size,
                              hipStream_t stream)
{
    const int*   sentence       = (const int*)  d_in[0];
    const int*   query_sentence = (const int*)  d_in[1];
    const float* query_idf      = (const float*)d_in[2];
    const float* emb            = (const float*)d_in[3];
    const float* ffw_w1         = (const float*)d_in[4];
    const float* ffw_b1         = (const float*)d_in[5];
    const float* ffw_w2         = (const float*)d_in[6];
    const float* ffw_b2         = (const float*)d_in[7];
    const float* gate_w         = (const float*)d_in[8];
    const float* out_w          = (const float*)d_in[9];
    const float* out_b          = (const float*)d_in[10];
    float* out = (float*)d_out;

    // workspace layout
    _Float16* dn = (_Float16*)d_ws;                         // B*TD*KPAD f16 (~168 MB)
    _Float16* qn = dn + (size_t)BATCH * TD * KPAD;          // B*TQ*KPAD f16 (~1.3 MB)
    float* ffw_vec = (float*)(qn + (size_t)BATCH * TQ * KPAD);  // B*TQ f32

    {
        const int nrows = BATCH * TD + BATCH * TQ;          // one wave per row
        const int blocks = (nrows + 7) / 8;                 // 8 waves / block
        gather_norm_kernel<<<blocks, 256, 0, stream>>>(
            sentence, query_sentence, emb, dn, qn);
    }
    gemm_hist_ffw_kernel<<<BATCH * 2, 256, 0, stream>>>(
        qn, dn, sentence, ffw_w1, ffw_b1, ffw_w2, ffw_b2, ffw_vec);
    final_kernel<<<BATCH, 32, 0, stream>>>(
        query_idf, query_sentence, ffw_vec, gate_w, out_w, out_b, out);
}
